// MixedAttention_46505905881416
// MI455X (gfx1250) — matrix operand order repaired and checked by reading, not compiled, not run on hardware
//
#include <hip/hip_runtime.h>
#include <hip/hip_bf16.h>

typedef __attribute__((ext_vector_type(16))) _Float16 v16h;
typedef __attribute__((ext_vector_type(8)))  _Float16 v8h;
typedef __attribute__((ext_vector_type(8)))  float    v8f;

// ---------------- problem constants ----------------
constexpr int BB  = 4;
constexpr int DD  = 512;
constexpr int TQ  = 22 * 64;       // 1408
constexpr int HH  = 8;
constexpr int DH  = 64;            // DD / HH
constexpr int KV  = 2 * TQ;        // 2816
constexpr int MROWS = BB * TQ;     // 5632

// ---------------- workspace layout (element offsets, f16) ----------------
constexpr size_t W_EL   = (size_t)DD * DD;            // 262144
constexpr size_t XN_EL  = (size_t)MROWS * DD;         // 2,883,584
constexpr size_t K_EL   = (size_t)BB * KV * DD;       // 5,767,168
constexpr size_t O_WQ   = 0 * W_EL;
constexpr size_t O_WKC  = 1 * W_EL;
constexpr size_t O_WKS  = 2 * W_EL;
constexpr size_t O_WVC  = 3 * W_EL;
constexpr size_t O_WVS  = 4 * W_EL;
constexpr size_t O_WZ   = 5 * W_EL;
constexpr size_t O_XN   = 6 * W_EL;
constexpr size_t O_X2N  = O_XN  + XN_EL;
constexpr size_t O_Q    = O_X2N + XN_EL;
constexpr size_t O_K    = O_Q   + XN_EL;
constexpr size_t O_VT   = O_K   + K_EL;
constexpr size_t O_Y    = O_VT  + K_EL;
constexpr size_t HALF_TOTAL = O_Y + XN_EL;
constexpr size_t STATS_BYTE_OFF = ((HALF_TOTAL * 2 + 255) / 256) * 256;

// ---------------- WMMA helpers ----------------
static __device__ __forceinline__ v8f wmma_f16(v16h a, v16h b, v8f c) {
  return __builtin_amdgcn_wmma_f32_16x16x32_f16(false, a, false, b,
                                                (short)0, c, false, false);
}

// A fragment 16x32 f16: p already points at rowptr + k0 + (lane>>4)*16.
static __device__ __forceinline__ v16h load_afrag(const _Float16* p) {
  v8h lo = *(const v8h*)(p);
  v8h hi = *(const v8h*)(p + 8);
  v16h a;
#pragma unroll
  for (int i = 0; i < 8; ++i) { a[i] = lo[i]; a[8 + i] = hi[i]; }
  return a;
}

// ---------------- kernel 0: f32 -> f16 weight convert ----------------
__global__ void cvt_f16_kernel(const float* __restrict__ src,
                               _Float16* __restrict__ dst, int n) {
  int i = blockIdx.x * blockDim.x + threadIdx.x;
  if (i < n) dst[i] = (_Float16)src[i];
}

// ---------------- kernel 1: layernorm stats (mean / rstd per (b,t)) -------
__global__ void ln_stats_kernel(const float* __restrict__ x,
                                float* __restrict__ mean,
                                float* __restrict__ rstd, int rows) {
  int idx = blockIdx.x * blockDim.x + threadIdx.x;
  if (idx >= rows) return;
  int b = idx / TQ, t = idx % TQ;
  const float* p = x + (size_t)b * DD * TQ + t;
  float s = 0.f, sq = 0.f;
  for (int d = 0; d < DD; ++d) {
    float v = p[(size_t)d * TQ];
    s += v; sq += v * v;
  }
  float m = s * (1.0f / DD);
  float var = sq * (1.0f / DD) - m * m;
  mean[idx] = m;
  rstd[idx] = rsqrtf(var + 1e-6f);
}

// ---------------- kernel 2: normalize + transpose (B,D,Tq)->(B*Tq,D) f16 --
__global__ void ln_apply_kernel(const float* __restrict__ x,
                                const float* __restrict__ mean,
                                const float* __restrict__ rstd,
                                const float* __restrict__ w,
                                const float* __restrict__ bias,
                                _Float16* __restrict__ out) {
  __shared__ float tile[32][33];
  int t0 = blockIdx.x * 32, d0 = blockIdx.y * 32, b = blockIdx.z;
  int tx = threadIdx.x, ty = threadIdx.y;
#pragma unroll
  for (int i = 0; i < 4; ++i) {
    int d = d0 + ty + i * 8;
    tile[ty + i * 8][tx] = x[((size_t)b * DD + d) * TQ + t0 + tx];
  }
  __syncthreads();
#pragma unroll
  for (int i = 0; i < 4; ++i) {
    int t = t0 + ty + i * 8;
    int d = d0 + tx;
    int row = b * TQ + t;
    float v = tile[tx][ty + i * 8];
    out[(size_t)row * DD + d] =
        (_Float16)((v - mean[row]) * rstd[row] * w[d] + bias[d]);
  }
}

// ---------------- kernel 3: 32x64-per-wave WMMA GEMM ----------------------
// out[m, n] = sum_k A[m,k] * W[n,k] + bias[n]     (M x 512 @ 512 x 512)
// mode 0: f16 row-major out[orow*512 + n], orow = batch*rpbOut + partOff + lr
// mode 1: f16 transposed out[(batch*512 + n)*rpbOut + partOff + lr]   (for V)
// mode 2: f32 out[(batch*512+n)*TQ + lr] = x[same] + val                (final)
__global__ __launch_bounds__(128)
void gemm512_kernel(const _Float16* __restrict__ A,
                    const _Float16* __restrict__ W,
                    const float* __restrict__ bias,
                    _Float16* __restrict__ outH,
                    float* __restrict__ outF,
                    const float* __restrict__ xadd,
                    int Mtiles32, int mode, int partOff, int rpbIn, int rpbOut) {
  int wave = blockIdx.x * (blockDim.x >> 5) + (threadIdx.x >> 5);
  if (wave >= Mtiles32 * 8) return;
  int mtile = wave >> 3, nt64 = wave & 7;
  int lane = threadIdx.x & 31;
  int lidx = lane & 15, lhalf = lane >> 4;
  int m0 = mtile * 32, n0 = nt64 * 64;

  v8f acc[2][4] = {};
  const _Float16* arow0 = A + (size_t)(m0 + lidx) * DD + lhalf * 16;
  const _Float16* arow1 = A + (size_t)(m0 + 16 + lidx) * DD + lhalf * 16;
  for (int k0 = 0; k0 < DD; k0 += 32) {
    // prefetch the next k-step of the A stream (gfx1250 global_prefetch_b8)
    if (k0 + 32 < DD) {
      __builtin_prefetch(arow0 + k0 + 32, 0, 1);
      __builtin_prefetch(arow1 + k0 + 32, 0, 1);
    }
    v16h a0 = load_afrag(arow0 + k0);
    v16h a1 = load_afrag(arow1 + k0);
#pragma unroll
    for (int t = 0; t < 4; ++t) {
      v16h b = *(const v16h*)(W + (size_t)(n0 + t * 16 + lidx) * DD +
                              k0 + lhalf * 16);
      acc[0][t] = wmma_f16(a0, b, acc[0][t]);
      acc[1][t] = wmma_f16(a1, b, acc[1][t]);
    }
  }
#pragma unroll
  for (int half = 0; half < 2; ++half) {
#pragma unroll
    for (int t = 0; t < 4; ++t) {
      int n = n0 + t * 16 + lidx;
      float bv = bias[n];
#pragma unroll
      for (int r = 0; r < 8; ++r) {
        int m = m0 + half * 16 + r + lhalf * 8;
        int bi = m / rpbIn, lr = m % rpbIn;
        float val = acc[half][t][r] + bv;
        if (mode == 0) {
          outH[(size_t)(bi * rpbOut + partOff + lr) * DD + n] = (_Float16)val;
        } else if (mode == 1) {
          outH[(size_t)(bi * DD + n) * rpbOut + partOff + lr] = (_Float16)val;
        } else {
          size_t idx = (size_t)(bi * DD + n) * TQ + lr;
          outF[idx] = xadd[idx] + val;
        }
      }
    }
  }
}

// ---------------- kernel 4: flash attention, 1 wave per (b,h,qtile) -------
__global__ __launch_bounds__(32)
void attn_kernel(const _Float16* __restrict__ qh,
                 const _Float16* __restrict__ kh,
                 const _Float16* __restrict__ vt,
                 _Float16* __restrict__ yh) {
  __shared__ float    sS[16][34];
  __shared__ _Float16 sP[16][32];
  __shared__ float    sAlpha[16];
  __shared__ float    sL[16];

  int qt = blockIdx.x, h = blockIdx.y, b = blockIdx.z;
  int lane = threadIdx.x;
  int lidx = lane & 15, lhalf = lane >> 4;

  // Q fragments: 16 queries x 64 dh -> two 16x32 A fragments (loaded once)
  const _Float16* qbase =
      qh + (size_t)(b * TQ + qt * 16 + lidx) * DD + h * DH;
  v16h aQ0 = load_afrag(qbase + 0  + lhalf * 16);
  v16h aQ1 = load_afrag(qbase + 32 + lhalf * 16);

  v8f acc[4] = {};
  float mRun = -__builtin_inff();
  float lRun = 0.0f;

  for (int c = 0; c < KV; c += 32) {
    // prefetch next key chunk's K rows and V columns one iteration ahead
    if (c + 32 < KV) {
      __builtin_prefetch(
          kh + (size_t)(b * KV + c + 32 + lane) * DD + h * DH, 0, 1);
      __builtin_prefetch(
          vt + (size_t)(b * DD + h * DH + (lane & 15)) * KV + c + 32, 0, 1);
    }
    // ---- S tiles: (16 queries) x (32 keys), dh contraction via 2 WMMAs ---
    v8f s0 = {}, s1 = {};
    {
      const _Float16* kb0 =
          kh + (size_t)(b * KV + c + lidx) * DD + h * DH;
      s0 = wmma_f16(aQ0, *(const v16h*)(kb0 + 0  + lhalf * 16), s0);
      s0 = wmma_f16(aQ1, *(const v16h*)(kb0 + 32 + lhalf * 16), s0);
      const _Float16* kb1 =
          kh + (size_t)(b * KV + c + 16 + lidx) * DD + h * DH;
      s1 = wmma_f16(aQ0, *(const v16h*)(kb1 + 0  + lhalf * 16), s1);
      s1 = wmma_f16(aQ1, *(const v16h*)(kb1 + 32 + lhalf * 16), s1);
    }
#pragma unroll
    for (int r = 0; r < 8; ++r) {
      sS[r + lhalf * 8][lidx]      = s0[r];
      sS[r + lhalf * 8][16 + lidx] = s1[r];
    }
    __syncthreads();

    // ---- online softmax, one lane per query row (lanes 0..15) ----
    if (lane < 16) {
      float mx = -__builtin_inff();
      for (int j = 0; j < 32; ++j) mx = fmaxf(mx, sS[lane][j]);
      float mNew = fmaxf(mRun, mx);
      float alpha = __expf(mRun - mNew);
      float sum = 0.0f;
      for (int j = 0; j < 32; ++j) {
        float p = __expf(sS[lane][j] - mNew);
        sP[lane][j] = (_Float16)p;
        sum += p;
      }
      lRun = lRun * alpha + sum;
      mRun = mNew;
      sAlpha[lane] = alpha;
    }
    __syncthreads();

    // ---- rescale accumulators, then y += P (16x32) @ V (32x64) ----
    float al[8];
#pragma unroll
    for (int r = 0; r < 8; ++r) al[r] = sAlpha[r + lhalf * 8];
#pragma unroll
    for (int t = 0; t < 4; ++t)
#pragma unroll
      for (int r = 0; r < 8; ++r) acc[t][r] *= al[r];

    v16h aP;
    {
      v8h lo = *(const v8h*)&sP[lidx][lhalf * 16];
      v8h hi = *(const v8h*)&sP[lidx][lhalf * 16 + 8];
#pragma unroll
      for (int i = 0; i < 8; ++i) { aP[i] = lo[i]; aP[8 + i] = hi[i]; }
    }
#pragma unroll
    for (int t = 0; t < 4; ++t) {
      const _Float16* vb =
          vt + (size_t)(b * DD + h * DH + t * 16 + lidx) * KV + c + lhalf * 16;
      acc[t] = wmma_f16(aP, *(const v16h*)vb, acc[t]);
    }
    __syncthreads();
  }

  if (lane < 16) sL[lane] = lRun;
  __syncthreads();
  float li[8];
#pragma unroll
  for (int r = 0; r < 8; ++r) li[r] = 1.0f / sL[r + lhalf * 8];
#pragma unroll
  for (int t = 0; t < 4; ++t)
#pragma unroll
    for (int r = 0; r < 8; ++r)
      yh[(size_t)(b * TQ + qt * 16 + r + lhalf * 8) * DD +
         h * DH + t * 16 + lidx] = (_Float16)(acc[t][r] * li[r]);
}

// ---------------- host side ----------------
extern "C" void kernel_launch(void* const* d_in, const int* in_sizes, int n_in,
                              void* d_out, int out_size, void* d_ws,
                              size_t ws_size, hipStream_t stream) {
  const float* x    = (const float*)d_in[0];
  const float* x2   = (const float*)d_in[1];
  const float* lnsw = (const float*)d_in[2];
  const float* lnsb = (const float*)d_in[3];
  const float* lncw = (const float*)d_in[4];
  const float* lncb = (const float*)d_in[5];
  const float* Wq  = (const float*)d_in[6];   const float* bq  = (const float*)d_in[7];
  const float* Wkc = (const float*)d_in[8];   const float* bkc = (const float*)d_in[9];
  const float* Wks = (const float*)d_in[10];  const float* bks = (const float*)d_in[11];
  const float* Wvc = (const float*)d_in[12];  const float* bvc = (const float*)d_in[13];
  const float* Wvs = (const float*)d_in[14];  const float* bvs = (const float*)d_in[15];
  const float* Wz  = (const float*)d_in[16];  const float* bz  = (const float*)d_in[17];

  _Float16* h = (_Float16*)d_ws;
  _Float16 *wqH = h + O_WQ, *wkcH = h + O_WKC, *wksH = h + O_WKS;
  _Float16 *wvcH = h + O_WVC, *wvsH = h + O_WVS, *wzH = h + O_WZ;
  _Float16 *xn = h + O_XN, *x2n = h + O_X2N;
  _Float16 *qH = h + O_Q, *kH = h + O_K, *vT = h + O_VT, *yH = h + O_Y;
  float* stats = (float*)((char*)d_ws + STATS_BYTE_OFF);
  float *meanX = stats, *rstdX = stats + MROWS;
  float *meanX2 = stats + 2 * MROWS, *rstdX2 = stats + 3 * MROWS;

  // 0: weights -> f16 (each 512x512 -> 1024 blocks of 256)
  const int wn = DD * DD;
  cvt_f16_kernel<<<wn / 256, 256, 0, stream>>>(Wq,  wqH,  wn);
  cvt_f16_kernel<<<wn / 256, 256, 0, stream>>>(Wkc, wkcH, wn);
  cvt_f16_kernel<<<wn / 256, 256, 0, stream>>>(Wks, wksH, wn);
  cvt_f16_kernel<<<wn / 256, 256, 0, stream>>>(Wvc, wvcH, wn);
  cvt_f16_kernel<<<wn / 256, 256, 0, stream>>>(Wvs, wvsH, wn);
  cvt_f16_kernel<<<wn / 256, 256, 0, stream>>>(Wz,  wzH,  wn);

  // 1: layernorm stats
  int sb = (MROWS + 255) / 256;
  ln_stats_kernel<<<sb, 256, 0, stream>>>(x,  meanX,  rstdX,  MROWS);
  ln_stats_kernel<<<sb, 256, 0, stream>>>(x2, meanX2, rstdX2, MROWS);

  // 2: normalize + transpose to (B*Tq, D) f16
  dim3 lgrid(TQ / 32, DD / 32, BB), lblk(32, 8);
  ln_apply_kernel<<<lgrid, lblk, 0, stream>>>(x,  meanX,  rstdX,  lnsw, lnsb, xn);
  ln_apply_kernel<<<lgrid, lblk, 0, stream>>>(x2, meanX2, rstdX2, lncw, lncb, x2n);

  // 3: projections. 176 m32-tiles * 8 n64-tiles = 1408 waves -> 352 blocks
  const int Mtiles32 = MROWS / 32;             // 176
  const int gblocks = (Mtiles32 * 8) / 4;      // 352
  gemm512_kernel<<<gblocks, 128, 0, stream>>>(xn,  wqH,  bq,  qH, nullptr, nullptr,
                                              Mtiles32, 0, 0,  TQ, TQ);   // Q
  gemm512_kernel<<<gblocks, 128, 0, stream>>>(x2n, wkcH, bkc, kH, nullptr, nullptr,
                                              Mtiles32, 0, 0,  TQ, KV);   // K cross
  gemm512_kernel<<<gblocks, 128, 0, stream>>>(xn,  wksH, bks, kH, nullptr, nullptr,
                                              Mtiles32, 0, TQ, TQ, KV);   // K self
  gemm512_kernel<<<gblocks, 128, 0, stream>>>(x2n, wvcH, bvc, vT, nullptr, nullptr,
                                              Mtiles32, 1, 0,  TQ, KV);   // V cross (transposed)
  gemm512_kernel<<<gblocks, 128, 0, stream>>>(xn,  wvsH, bvs, vT, nullptr, nullptr,
                                              Mtiles32, 1, TQ, TQ, KV);   // V self  (transposed)

  // 4: flash attention
  attn_kernel<<<dim3(TQ / 16, HH, BB), 32, 0, stream>>>(qH, kH, vT, yH);

  // 5: out = x + y @ Wz^T + bz, written back in (B, D, J, T) f32 layout
  gemm512_kernel<<<gblocks, 128, 0, stream>>>(yH, wzH, bz, nullptr, (float*)d_out,
                                              x, Mtiles32, 2, 0, TQ, TQ);
}
